// CQAttention_34213709480321
// MI455X (gfx1250) — matrix operand order repaired and checked by reading, not compiled, not run on hardware
//
#include <hip/hip_runtime.h>
#include <stdint.h>

// Problem constants
#define B_   64
#define D_   256
#define LC_  1024
#define LQ_  128

typedef __attribute__((ext_vector_type(16))) __bf16 v16bf;
typedef __attribute__((ext_vector_type(8)))  float  v8f;
typedef __attribute__((ext_vector_type(8)))  unsigned int v8u;

// ---------- bf16 helpers (round-to-nearest-even, packed pairs) ----------
__device__ __forceinline__ uint32_t f2bf1(float f) {
  uint32_t u = __float_as_uint(f);
  return (u + 0x7fffu + ((u >> 16) & 1u)) >> 16;
}
__device__ __forceinline__ uint32_t pack_bf2(float lo, float hi) {
  return f2bf1(lo) | (f2bf1(hi) << 16);
}

// ---------- WMMA fragment loaders ----------
// LDS layouts: rows of 16 uint32 (each uint32 = packed bf16 pair K=2k2, 2k2+1).
// A (16x32 MxK): lane<16: M=lane, K=0..7 & 16..23; lane>=16: M=lane-16, K=8..15 & 24..31.
__device__ __forceinline__ v16bf frag_a_load(const uint32_t* row16, int lane) {
  const int kb2 = (lane >= 16) ? 4 : 0;
  v8u r;
  r[0] = row16[kb2 + 0];  r[1] = row16[kb2 + 1];
  r[2] = row16[kb2 + 2];  r[3] = row16[kb2 + 3];
  r[4] = row16[kb2 + 8];  r[5] = row16[kb2 + 9];
  r[6] = row16[kb2 + 10]; r[7] = row16[kb2 + 11];
  return __builtin_bit_cast(v16bf, r);
}
// B (32x16 KxN): lane<16: N=lane, K=0..7 & 16..23; lane>=16: N=lane-16, K=8..15 & 24..31.
__device__ __forceinline__ v16bf frag_b_load(const uint32_t* row16, int lane) {
  const int kb2 = (lane >= 16) ? 4 : 0;
  v8u r;
  r[0] = row16[kb2 + 0]; r[1] = row16[kb2 + 1];
  r[2] = row16[kb2 + 2]; r[3] = row16[kb2 + 3];
  r[4] = row16[kb2 + 8]; r[5] = row16[kb2 + 9];
  r[6] = row16[kb2 + 10]; r[7] = row16[kb2 + 11];
  return __builtin_bit_cast(v16bf, r);
}
#define WMMA_BF16(A, Bm, Cacc) \
  __builtin_amdgcn_wmma_f32_16x16x32_bf16(false, (A), false, (Bm), (short)0, (Cacc), false, false)

// ---------- Faithful JAX dropout: threefry2x32-20, key(42) = (0, 42) ----------
__device__ __forceinline__ float jax_drop_scale(uint32_t idx) {
  const uint32_t N    = 64u * 1024u * 1024u;   // out elements
  const uint32_t half = N >> 1;
  const bool lo = idx < half;
  uint32_t c0 = lo ? idx : (idx - half);
  uint32_t c1 = c0 + half;
  const uint32_t k0 = 0u, k1 = 42u;
  const uint32_t k2 = k0 ^ k1 ^ 0x1BD11BDAu;
  uint32_t x0 = c0 + k0, x1 = c1 + k1;
#define TF_R(r) { x0 += x1; x1 = (x1 << (r)) | (x1 >> (32 - (r))); x1 ^= x0; }
  TF_R(13) TF_R(15) TF_R(26) TF_R(6)   x0 += k1; x1 += k2 + 1u;
  TF_R(17) TF_R(29) TF_R(16) TF_R(24)  x0 += k2; x1 += k0 + 2u;
  TF_R(13) TF_R(15) TF_R(26) TF_R(6)   x0 += k0; x1 += k1 + 3u;
  TF_R(17) TF_R(29) TF_R(16) TF_R(24)  x0 += k1; x1 += k2 + 4u;
  TF_R(13) TF_R(15) TF_R(26) TF_R(6)   x0 += k2; x1 += k0 + 5u;
#undef TF_R
  uint32_t bits = lo ? x0 : x1;
  float u = __uint_as_float((bits >> 9) | 0x3f800000u) - 1.0f;
  return (u < 0.9f) ? (1.0f / 0.9f) : 0.0f;
}

// ============ K1: S = (wm*C)^T Q + sc + sq; fused sc/sq; fused row softmax ===
// Block: 128 i x 128 j, 8 waves (wave = i-subtile, loops 8 j-subtiles).
// Writes S (fp32) and Erow = exp(S-rowmax)/rowsum as packed-bf16 pairs.
__global__ __launch_bounds__(256) void k1_scores(
    const float* __restrict__ C, const float* __restrict__ Q,
    const float* __restrict__ W, float* __restrict__ S,
    uint32_t* __restrict__ Erow) {
  __shared__ uint32_t ldsA[128 * 16];   // (wm*C) tile [i_local][k2]
  __shared__ uint32_t ldsB[128 * 16];   // Q tile      [j][k2]
  __shared__ float    ldsSC[128];
  __shared__ float    ldsSQ[128];
  const int b  = blockIdx.y;
  const int i0 = blockIdx.x * 128;
  const int tid = threadIdx.x, lane = tid & 31, wave = tid >> 5;
  const float* wq = W + (size_t)b * 3 * D_;
  const float* wc = wq + D_;
  const float* wm = wq + 2 * D_;

  if (tid < 128) { ldsSC[tid] = 0.f; ldsSQ[tid] = 0.f; }

  const v8f vzero = {0.f,0.f,0.f,0.f,0.f,0.f,0.f,0.f};
  v8f acc[8] = {vzero, vzero, vzero, vzero, vzero, vzero, vzero, vzero};
  float scp = 0.f, sqp = 0.f;

  for (int d0 = 0; d0 < D_; d0 += 32) {
#pragma unroll
    for (int it = 0; it < 8; ++it) {          // stage A: (wm*C) [128 i][32 k]
      const int idx = tid + 256 * it;
      const int il = idx & 127, k2 = idx >> 7;      // il fixed = tid&127
      const int d = d0 + 2 * k2;
      const float c0 = C[((size_t)b * D_ + d) * LC_ + i0 + il];
      const float c1 = C[((size_t)b * D_ + d + 1) * LC_ + i0 + il];
      ldsA[il * 16 + k2] = pack_bf2(wm[d] * c0, wm[d + 1] * c1);
      scp += wc[d] * c0 + wc[d + 1] * c1;
    }
#pragma unroll
    for (int it = 0; it < 8; ++it) {          // stage B: Q [32 k][128 j]
      const int idx = tid + 256 * it;
      const int j = idx & 127, k2 = idx >> 7;       // j fixed = tid&127
      const int d = d0 + 2 * k2;
      const float q0 = Q[((size_t)b * D_ + d) * LQ_ + j];
      const float q1 = Q[((size_t)b * D_ + d + 1) * LQ_ + j];
      ldsB[j * 16 + k2] = pack_bf2(q0, q1);
      sqp += wq[d] * q0 + wq[d + 1] * q1;
    }
    __syncthreads();
    const v16bf a = frag_a_load(&ldsA[(wave * 16 + (lane & 15)) * 16], lane);
#pragma unroll
    for (int js = 0; js < 8; ++js) {
      const v16bf bb = frag_b_load(&ldsB[(js * 16 + (lane & 15)) * 16], lane);
      acc[js] = WMMA_BF16(a, bb, acc[js]);
    }
    __syncthreads();
  }
  atomicAdd(&ldsSC[tid & 127], scp);   // ds_add_f32
  atomicAdd(&ldsSQ[tid & 127], sqp);
  __syncthreads();

  const int n = lane & 15;
  const int mofs = (lane >= 16) ? 8 : 0;
  float sqv[8];
#pragma unroll
  for (int js = 0; js < 8; ++js) sqv[js] = ldsSQ[js * 16 + n];

#pragma unroll
  for (int r = 0; r < 8; ++r) {
    const int i = i0 + wave * 16 + r + mofs;
    const float scv = ldsSC[wave * 16 + r + mofs];
    float v[8];
#pragma unroll
    for (int js = 0; js < 8; ++js) v[js] = acc[js][r] + scv + sqv[js];
    // row softmax stats: rows live entirely in one 16-lane half of this wave
    float m = v[0];
#pragma unroll
    for (int js = 1; js < 8; ++js) m = fmaxf(m, v[js]);
#pragma unroll
    for (int off = 1; off <= 8; off <<= 1) m = fmaxf(m, __shfl_xor(m, off, 32));
    float e[8], s = 0.f;
#pragma unroll
    for (int js = 0; js < 8; ++js) { e[js] = __expf(v[js] - m); s += e[js]; }
#pragma unroll
    for (int off = 1; off <= 8; off <<= 1) s += __shfl_xor(s, off, 32);
    const float inv = 1.0f / s;
#pragma unroll
    for (int js = 0; js < 8; ++js) {
      S[((size_t)b * LC_ + i) * LQ_ + js * 16 + n] = v[js];
      const float ev = e[js] * inv;
      const float po = __shfl_xor(ev, 1, 32);
      if ((n & 1) == 0)
        Erow[((size_t)b * LC_ + i) * 64 + js * 8 + (n >> 1)] = pack_bf2(ev, po);
    }
  }
}

// ====== K2: per-column (axis i) partial online softmax stats (8 chunks) ======
__global__ __launch_bounds__(128) void k2_col_part(
    const float* __restrict__ S, float* __restrict__ pmax, float* __restrict__ psum) {
  const int b = blockIdx.x, chunk = blockIdx.y, j = threadIdx.x;
  const float* p = S + ((size_t)b * LC_ + chunk * 128) * LQ_ + j;
  float m = -3.0e38f, s = 0.f;
  for (int i = 0; i < 128; ++i) {
    const float x = p[(size_t)i * LQ_];
    const float nm = fmaxf(m, x);
    s = s * __expf(m - nm) + __expf(x - nm);
    m = nm;
  }
  pmax[((size_t)chunk * B_ + b) * LQ_ + j] = m;
  psum[((size_t)chunk * B_ + b) * LQ_ + j] = s;
}
__global__ __launch_bounds__(128) void k2_col_combine(
    const float* __restrict__ pmax, const float* __restrict__ psum,
    float* __restrict__ colmax, float* __restrict__ colsum) {
  const int b = blockIdx.x, j = threadIdx.x;
  float m = -3.0e38f;
#pragma unroll
  for (int c = 0; c < 8; ++c) m = fmaxf(m, pmax[((size_t)c * B_ + b) * LQ_ + j]);
  float s = 0.f;
#pragma unroll
  for (int c = 0; c < 8; ++c)
    s += psum[((size_t)c * B_ + b) * LQ_ + j] *
         __expf(pmax[((size_t)c * B_ + b) * LQ_ + j] - m);
  colmax[b * LQ_ + j] = m;
  colsum[b * LQ_ + j] = s;
}

// ======== K3: M[b,d,j] = (sum_i C[d,i]*exp(S[i,j]-colmax[j])) / colsum[j] ====
// One block per batch: 256 d x 128 j; wave owns 2 d-subtiles (16 accumulators).
__global__ __launch_bounds__(256) void k3_M(
    const float* __restrict__ C, const float* __restrict__ S,
    const float* __restrict__ colmax, const float* __restrict__ colsum,
    float* __restrict__ Mout) {
  __shared__ uint32_t ldsA[256 * 16];   // C tile [d][k2]
  __shared__ uint32_t ldsB[128 * 16];   // exp(S-colmax)^T tile [j][k2]
  const int b = blockIdx.x;
  const int tid = threadIdx.x, lane = tid & 31, wave = tid >> 5;

  const v8f vzero = {0.f,0.f,0.f,0.f,0.f,0.f,0.f,0.f};
  v8f acc[16];
#pragma unroll
  for (int t = 0; t < 16; ++t) acc[t] = vzero;

  for (int i0 = 0; i0 < LC_; i0 += 32) {
#pragma unroll
    for (int it = 0; it < 16; ++it) {         // stage A: C[0..255][i0..i0+31]
      const int idx = tid + 256 * it;
      const int k2 = idx & 15, dl = idx >> 4;
      const float2 cv = *(const float2*)&C[((size_t)b * D_ + dl) * LC_ + i0 + 2 * k2];
      ldsA[dl * 16 + k2] = pack_bf2(cv.x, cv.y);
    }
#pragma unroll
    for (int it = 0; it < 8; ++it) {          // stage B: exp(S - colmax)^T
      const int idx = tid + 256 * it;
      const int j = idx & 127, k2 = idx >> 7;
      const float cm = colmax[b * LQ_ + j];
      const float s0 = S[((size_t)b * LC_ + i0 + 2 * k2) * LQ_ + j];
      const float s1 = S[((size_t)b * LC_ + i0 + 2 * k2 + 1) * LQ_ + j];
      ldsB[j * 16 + k2] = pack_bf2(__expf(s0 - cm), __expf(s1 - cm));
    }
    __syncthreads();
    const v16bf a0 = frag_a_load(&ldsA[(wave * 16 + (lane & 15)) * 16], lane);
    const v16bf a1 = frag_a_load(&ldsA[((128 + wave * 16) + (lane & 15)) * 16], lane);
#pragma unroll
    for (int js = 0; js < 8; ++js) {
      const v16bf bb = frag_b_load(&ldsB[(js * 16 + (lane & 15)) * 16], lane);
      acc[js]     = WMMA_BF16(a0, bb, acc[js]);
      acc[8 + js] = WMMA_BF16(a1, bb, acc[8 + js]);
    }
    __syncthreads();
  }
  const int mofs = (lane >= 16) ? 8 : 0;
#pragma unroll
  for (int js = 0; js < 8; ++js) {
    const int j = js * 16 + (lane & 15);
    const float inv = 1.0f / colsum[b * LQ_ + j];
#pragma unroll
    for (int r = 0; r < 8; ++r) {
      const int d = wave * 16 + r + mofs;
      Mout[((size_t)b * D_ + d) * LQ_ + j]       = acc[js][r] * inv;
      Mout[((size_t)b * D_ + d + 128) * LQ_ + j] = acc[8 + js][r] * inv;
    }
  }
}

// == K4: [A;Bmat] = [Q;M] * Erow^T (Erow pre-normalized); emit all 4 regions ==
// Block: 128 d' x 128 i, K = j (4 steps of 32). Grid y picks d' quarter.
// Stage B is a raw copy of packed bf16 -> use CDNA5 async LDS copy.
__global__ __launch_bounds__(256) void k4_out(
    const float* __restrict__ C, const float* __restrict__ Q,
    const uint32_t* __restrict__ Erow, const float* __restrict__ Mws,
    float* __restrict__ out) {
  __shared__ uint32_t ldsA[128 * 16];   // G tile [d_local][k2]
  __shared__ uint32_t ldsB[128 * 16];   // Erow tile [i_local][k2]
  const int b   = blockIdx.z;
  const int dp0 = blockIdx.y * 128;     // 0,128 -> A path; 256,384 -> Bmat path
  const int i0  = blockIdx.x * 128;
  const int tid = threadIdx.x, lane = tid & 31, wave = tid >> 5;
  const bool isA = (dp0 < 256);
  const float* G = isA ? (Q   + ((size_t)b * D_ + dp0)       * LQ_)
                       : (Mws + ((size_t)b * D_ + dp0 - 256) * LQ_);

  const v8f vzero = {0.f,0.f,0.f,0.f,0.f,0.f,0.f,0.f};
  v8f acc[8] = {vzero, vzero, vzero, vzero, vzero, vzero, vzero, vzero};

  for (int j0 = 0; j0 < LQ_; j0 += 32) {
    // stage B: async raw copy of 8KB packed-bf16 Erow tile (16B chunks)
#pragma unroll
    for (int it = 0; it < 2; ++it) {
      const int c = tid + 256 * it;           // 512 x 16B
      const int il = c >> 2, part = c & 3;
      const uint32_t laddr = (uint32_t)(uintptr_t)&ldsB[il * 16 + part * 4];
      const uint64_t gaddr =
          (uint64_t)(uintptr_t)&Erow[((size_t)b * LC_ + i0 + il) * 64 + (j0 >> 1) + part * 4];
      asm volatile("global_load_async_to_lds_b128 %0, %1, off"
                   :: "v"(laddr), "v"(gaddr) : "memory");
    }
#pragma unroll
    for (int it = 0; it < 8; ++it) {          // stage A: G[dl][j0+2k2..+1] -> bf16
      const int idx = tid + 256 * it;
      const int k2 = idx & 15, dl = idx >> 4;
      const float2 g = *(const float2*)&G[(size_t)dl * LQ_ + j0 + 2 * k2];
      ldsA[dl * 16 + k2] = pack_bf2(g.x, g.y);
    }
    asm volatile("s_wait_asynccnt 0x0" ::: "memory");
    __syncthreads();
    const v16bf a = frag_a_load(&ldsA[(wave * 16 + (lane & 15)) * 16], lane);
#pragma unroll
    for (int is = 0; is < 8; ++is) {
      const v16bf bb = frag_b_load(&ldsB[(is * 16 + (lane & 15)) * 16], lane);
      acc[is] = WMMA_BF16(a, bb, acc[is]);
    }
    __syncthreads();
  }
  const int mofs = (lane >= 16) ? 8 : 0;
#pragma unroll
  for (int is = 0; is < 8; ++is) {
    const int i = i0 + is * 16 + (lane & 15);
#pragma unroll
    for (int r = 0; r < 8; ++r) {
      const int dl  = wave * 16 + r + mofs;
      const float v = acc[is][r];             // Erow already /rowsum
      if (isA) {
        const int d = dp0 + dl;
        const float cv = C[((size_t)b * D_ + d) * LC_ + i];
        const uint32_t o0 = ((uint32_t)b * 1024u + (uint32_t)d)        * LC_ + i; // C
        const uint32_t o1 = ((uint32_t)b * 1024u + 256u + (uint32_t)d) * LC_ + i; // A
        const uint32_t o2 = ((uint32_t)b * 1024u + 512u + (uint32_t)d) * LC_ + i; // C*A
        out[o0] = cv      * jax_drop_scale(o0);
        out[o1] = v       * jax_drop_scale(o1);
        out[o2] = cv * v  * jax_drop_scale(o2);
      } else {
        const int d = dp0 - 256 + dl;
        const float cv = C[((size_t)b * D_ + d) * LC_ + i];
        const uint32_t o3 = ((uint32_t)b * 1024u + 768u + (uint32_t)d) * LC_ + i; // C*Bmat
        out[o3] = cv * v * jax_drop_scale(o3);
      }
    }
  }
}

// ============================== launcher =====================================
extern "C" void kernel_launch(void* const* d_in, const int* in_sizes, int n_in,
                              void* d_out, int out_size, void* d_ws, size_t ws_size,
                              hipStream_t stream) {
  (void)in_sizes; (void)n_in; (void)out_size; (void)ws_size;
  const float* C = (const float*)d_in[0];
  const float* Q = (const float*)d_in[1];
  const float* W = (const float*)d_in[2];
  float* out = (float*)d_out;

  char* ws = (char*)d_ws;
  float*    S      = (float*)(ws);                    // 33,554,432 B
  float*    Mws    = (float*)(ws + 33554432u);        //  8,388,608 B
  uint32_t* Erow   = (uint32_t*)(ws + 41943040u);     // 16,777,216 B (packed bf16)
  float*    colmax = (float*)(ws + 58720256u);        //     32,768 B
  float*    colsum = (float*)(ws + 58753024u);        //     32,768 B
  float*    pcmax  = (float*)(ws + 58785792u);        //    262,144 B
  float*    pcsum  = (float*)(ws + 59047936u);        //    262,144 B

  k1_scores<<<dim3(LC_ / 128, B_), 256, 0, stream>>>(C, Q, W, S, Erow);
  k2_col_part<<<dim3(B_, 8), 128, 0, stream>>>(S, pcmax, pcsum);
  k2_col_combine<<<B_, 128, 0, stream>>>(pcmax, pcsum, colmax, colsum);
  k3_M<<<B_, 256, 0, stream>>>(C, S, colmax, colsum, Mws);
  k4_out<<<dim3(LC_ / 128, 4, B_), 256, 0, stream>>>(C, Q, Erow, Mws, out);
}
